// AttentionBlock_83176336654666
// MI455X (gfx1250) — matrix operand order repaired and checked by reading, not compiled, not run on hardware
//
#include <hip/hip_runtime.h>
#include <hip/hip_bf16.h>

// ---------------------------------------------------------------------------
// Channel-attention block for MI455X (gfx1250, wave32, WMMA f16->f32).
// Pipeline: qkv_proj -> sim split-K (+f32 atomics) -> softmax -> attn@V
//           -> out_proj (+bias).  No LDS staging: WMMA fragments are gathered
//           directly from global with b128 vector loads where the contraction
//           dim is contiguous; D tiles stored as packed v8h/v8f along n.
// ---------------------------------------------------------------------------

typedef _Float16 half_t;
typedef __attribute__((ext_vector_type(16))) _Float16 v16h;
typedef __attribute__((ext_vector_type(8)))  _Float16 v8h;
typedef __attribute__((ext_vector_type(8)))  float    v8f;

#define NSPATIAL 16384   // 128*128
#define DH       64
#define QD       256     // HEADS * c(32)
#define INNER    512     // HEADS * DH
#define GOUT     1536    // q(512) + k(512) + v(512)
#define NBP      2       // b(16) / HEADS

static __device__ __forceinline__ v8f wmma_f16(v16h a, v16h b, v8f c) {
  // v_wmma_f32_16x16x32_f16: D = A(16x32,f16) x B(32x16,f16) + C(16x16,f32)
  return __builtin_amdgcn_wmma_f32_16x16x32_f16(
      false, a, false, b, (short)0, c, false, false);
}

static __device__ __forceinline__ v16h cat16(v8h lo, v8h hi) {
  return __builtin_shufflevector(lo, hi, 0, 1, 2, 3, 4, 5, 6, 7,
                                         8, 9, 10, 11, 12, 13, 14, 15);
}

// Fragment geometry per lane (wave32):
//  A 16x32: row m = l&15; K run {kb16..kb16+15} -> slots 0..15 (as B).
//  B 32x16: col n = l&15; K run {kb16..kb16+15} -> slots 0..15, kb16=16*(l>=16).
//  C/D 16x16: col = l&15; row = r + 8*(l>=16) for accumulator element r.

// ---------------------------------------------------------------------------
// Kernel 1: fused Q/K/V projection.
// A[n][ch] = x[bp][ch][n] (f32, contiguous along n -> lane-coalesced gathers),
// B[ch][o] = G[o][ch], G=[Wq;Wkv] (rows contiguous in ch -> float4 loads).
// Output transposed f16: qkv[bp][o][n]  (o<512:q, <1024:k, <1536:v).
// grid (N/128, 1536/64, 2), block 256 (8 waves, 16 n-rows x 64 o each).
// ---------------------------------------------------------------------------
__global__ __launch_bounds__(256) void qkv_proj_kernel(
    const float* __restrict__ x, const float* __restrict__ Wq,
    const float* __restrict__ Wkv, half_t* __restrict__ qkv) {
  const int n0 = blockIdx.x * 128;
  const int o0 = blockIdx.y * 64;
  const int bp = blockIdx.z;
  const int tid = threadIdx.x, wave = tid >> 5, l = tid & 31;
  const int m = l & 15, col = l & 15;
  const int kb8 = (l >> 4) << 3, kb16 = (l >> 4) << 4, rbase = kb8;
  const float* xb = x + (size_t)bp * 256 * NSPATIAL;
  const int nrow = n0 + wave * 16 + m;

  v8f acc[4];
#pragma unroll
  for (int j = 0; j < 4; ++j)
#pragma unroll
    for (int t = 0; t < 8; ++t) acc[j][t] = 0.0f;

  for (int kc = 0; kc < 256; kc += 32) {
    if (kc + 32 < 256)  // pull next channel chunk toward GL2
      __builtin_prefetch(&xb[(size_t)(kc + 32 + (tid >> 3)) * NSPATIAL + n0], 0, 0);
    // A fragment: 16 lane-coalesced b32 loads + cvt (x is n-contiguous).
    v16h a;
#pragma unroll
    for (int v = 0; v < 8; ++v) {
      const int k = kc + kb16 + (v << 1);
      a[2 * v]     = (half_t)xb[(size_t)k * NSPATIAL + nrow];
      a[2 * v + 1] = (half_t)xb[(size_t)(k + 1) * NSPATIAL + nrow];
    }
#pragma unroll
    for (int j = 0; j < 4; ++j) {
      const int o = o0 + j * 16 + col;
      const float* rp = ((o < INNER) ? (Wq + (size_t)o * QD)
                                     : (Wkv + (size_t)(o - INNER) * QD)) +
                        kc + kb16;
      const float4 w0 = ((const float4*)rp)[0];
      const float4 w1 = ((const float4*)rp)[1];
      const float4 w2 = ((const float4*)rp)[2];
      const float4 w3 = ((const float4*)rp)[3];
      v16h b;
      b[0] = (half_t)w0.x; b[1] = (half_t)w0.y; b[2] = (half_t)w0.z; b[3] = (half_t)w0.w;
      b[4] = (half_t)w1.x; b[5] = (half_t)w1.y; b[6] = (half_t)w1.z; b[7] = (half_t)w1.w;
      b[8] = (half_t)w2.x; b[9] = (half_t)w2.y; b[10] = (half_t)w2.z; b[11] = (half_t)w2.w;
      b[12] = (half_t)w3.x; b[13] = (half_t)w3.y; b[14] = (half_t)w3.z; b[15] = (half_t)w3.w;
      acc[j] = wmma_f16(a, b, acc[j]);
    }
  }
  // Store transposed qkv[bp][o][n]: 8 consecutive n per lane -> one v8h store.
  half_t* qb = qkv + (size_t)bp * GOUT * NSPATIAL;
#pragma unroll
  for (int j = 0; j < 4; ++j) {
    v8h st;
#pragma unroll
    for (int r = 0; r < 8; ++r) st[r] = (half_t)acc[j][r];
    *(v8h*)(qb + (size_t)(o0 + j * 16 + col) * NSPATIAL +
            (n0 + wave * 16 + rbase)) = st;
  }
}

// ---------------------------------------------------------------------------
// Kernel 2a: zero the sim accumulator (16*64*64 f32).
// ---------------------------------------------------------------------------
__global__ __launch_bounds__(256) void zero_sim_kernel(float* __restrict__ sim) {
#pragma unroll
  for (int i = 0; i < 4; ++i)
    sim[blockIdx.x * 1024 + i * 256 + threadIdx.x] = 0.0f;
}

// ---------------------------------------------------------------------------
// Kernel 2b: sim partial = scale * Q Kᵀ, split-K over 16 chunks of 1024.
// q/k rows are n-contiguous -> both fragments are pure b128 loads.
// grid (16 chunks, 16 heads), block 256; wave owns 1 i-tile x 2 j-tiles.
// ---------------------------------------------------------------------------
__global__ __launch_bounds__(256) void sim_kernel(
    const half_t* __restrict__ qkv, float* __restrict__ sim) {
  const int head = blockIdx.y;
  const int bp = head >> 3, h = head & 7;
  const int n0 = blockIdx.x * 1024;
  const int tid = threadIdx.x, wave = tid >> 5, l = tid & 31;
  const int m = l & 15, col = l & 15;
  const int kb8 = (l >> 4) << 3, kb16 = (l >> 4) << 4, rbase = kb8;
  const int it = wave >> 1, jt0 = (wave & 1) * 2;
  const half_t* qb = qkv + ((size_t)bp * GOUT + h * DH) * NSPATIAL;
  const half_t* kmat = qkv + ((size_t)bp * GOUT + INNER + h * DH) * NSPATIAL;
  const half_t* arow  = qb + (size_t)(it * 16 + m) * NSPATIAL;
  const half_t* brow0 = kmat + (size_t)(jt0 * 16 + col) * NSPATIAL;
  const half_t* brow1 = kmat + (size_t)((jt0 + 1) * 16 + col) * NSPATIAL;

  v8f acc[2];
#pragma unroll
  for (int j = 0; j < 2; ++j)
#pragma unroll
    for (int t = 0; t < 8; ++t) acc[j][t] = 0.0f;

  for (int nc = n0; nc < n0 + 1024; nc += 32) {
    const v16h a = cat16(*(const v8h*)(arow + nc + kb16),
                         *(const v8h*)(arow + nc + kb16 + 8));
    const v16h b0 = cat16(*(const v8h*)(brow0 + nc + kb16),
                          *(const v8h*)(brow0 + nc + kb16 + 8));
    const v16h b1 = cat16(*(const v8h*)(brow1 + nc + kb16),
                          *(const v8h*)(brow1 + nc + kb16 + 8));
    acc[0] = wmma_f16(a, b0, acc[0]);
    acc[1] = wmma_f16(a, b1, acc[1]);
  }
  const float scale = 0.125f;  // DIM_HEAD^-0.5
  float* sb = sim + head * 4096;
#pragma unroll
  for (int jj = 0; jj < 2; ++jj) {
    const int j = (jt0 + jj) * 16 + col;
#pragma unroll
    for (int r = 0; r < 8; ++r) {
      const int i = it * 16 + r + rbase;
      atomicAdd(&sb[i * 64 + j], acc[jj][r] * scale);
    }
  }
}

// ---------------------------------------------------------------------------
// Kernel 3: row softmax over 64 columns; output f16 attn matrix.
// grid (16*64 rows), block 64.
// ---------------------------------------------------------------------------
__global__ __launch_bounds__(64) void softmax_kernel(
    const float* __restrict__ sim, half_t* __restrict__ attn) {
  __shared__ float red[64];
  const int row = blockIdx.x;
  const int j = threadIdx.x;
  const float s = sim[row * 64 + j];
  red[j] = s;
  __syncthreads();
  for (int off = 32; off > 0; off >>= 1) {
    if (j < off) red[j] = fmaxf(red[j], red[j + off]);
    __syncthreads();
  }
  const float mx = red[0];
  __syncthreads();
  const float e = __expf(s - mx);
  red[j] = e;
  __syncthreads();
  for (int off = 32; off > 0; off >>= 1) {
    if (j < off) red[j] += red[j + off];
    __syncthreads();
  }
  attn[row * 64 + j] = (half_t)(e * (1.0f / red[0]));
}

// ---------------------------------------------------------------------------
// Kernel 4: outᵀ = Vᵀ(16384x64) x attnᵀ(64x64)  (i.e. out[i][n]=Σ attn[i][j]v[j][n]).
// A[m=n][k=j] = v[j][n] (lane-coalesced b16 gathers), B[k=j][c=i] = attn[i][j]
// (rows contiguous in j -> b128 loads).  D: 8 consecutive n per lane -> v8h.
// grid (N/128, 16 heads), block 256; wave = 16 n-rows x 64 i.
// ---------------------------------------------------------------------------
__global__ __launch_bounds__(256) void attnv_kernel(
    const half_t* __restrict__ qkv, const half_t* __restrict__ attn,
    half_t* __restrict__ oT) {
  const int head = blockIdx.y;
  const int bp = head >> 3, h = head & 7;
  const int n0 = blockIdx.x * 128;
  const int tid = threadIdx.x, wave = tid >> 5, l = tid & 31;
  const int m = l & 15, col = l & 15;
  const int kb8 = (l >> 4) << 3, kb16 = (l >> 4) << 4, rbase = kb8;
  const half_t* vb = qkv + ((size_t)bp * GOUT + 2 * INNER + h * DH) * NSPATIAL;
  const half_t* ab = attn + head * 4096;
  const int nrow = n0 + wave * 16 + m;

  v8f acc[4];
#pragma unroll
  for (int j = 0; j < 4; ++j)
#pragma unroll
    for (int t = 0; t < 8; ++t) acc[j][t] = 0.0f;

#pragma unroll
  for (int ks = 0; ks < 2; ++ks) {
    v16h a;
#pragma unroll
    for (int v = 0; v < 8; ++v) {
      const int k = ks * 32 + kb16 + (v << 1);
      a[2 * v]     = vb[(size_t)k * NSPATIAL + nrow];
      a[2 * v + 1] = vb[(size_t)(k + 1) * NSPATIAL + nrow];
    }
#pragma unroll
    for (int j = 0; j < 4; ++j) {
      const half_t* br = ab + (size_t)(j * 16 + col) * 64 + ks * 32 + kb16;
      const v16h b = cat16(*(const v8h*)br, *(const v8h*)(br + 8));
      acc[j] = wmma_f16(a, b, acc[j]);
    }
  }
  half_t* ob = oT + (size_t)head * DH * NSPATIAL;
#pragma unroll
  for (int j = 0; j < 4; ++j) {
    v8h st;
#pragma unroll
    for (int r = 0; r < 8; ++r) st[r] = (half_t)acc[j][r];
    *(v8h*)(ob + (size_t)(j * 16 + col) * NSPATIAL +
            (n0 + wave * 16 + rbase)) = st;
  }
}

// ---------------------------------------------------------------------------
// Kernel 5: output projection + bias, K=512.
// A[n][o] = oT[bp*8+o/64][o%64][n] (lane-coalesced b16 gathers),
// B[o][cq] = Wout[cq][o] (rows contiguous in o -> float4 loads + cvt).
// Writes f32 out[bp][cq][n]; 8 consecutive n per lane -> 2x float4 stores.
// grid (N/128, 256/64, 2), block 256.
// ---------------------------------------------------------------------------
__global__ __launch_bounds__(256) void outproj_kernel(
    const half_t* __restrict__ oT, const float* __restrict__ Wout,
    const float* __restrict__ bout, float* __restrict__ out) {
  const int n0 = blockIdx.x * 128;
  const int c0 = blockIdx.y * 64;
  const int bp = blockIdx.z;
  const int tid = threadIdx.x, wave = tid >> 5, l = tid & 31;
  const int m = l & 15, col = l & 15;
  const int kb8 = (l >> 4) << 3, kb16 = (l >> 4) << 4, rbase = kb8;
  const int nrow = n0 + wave * 16 + m;

  v8f acc[4];
#pragma unroll
  for (int j = 0; j < 4; ++j)
#pragma unroll
    for (int t = 0; t < 8; ++t) acc[j][t] = 0.0f;

  for (int kc = 0; kc < INNER; kc += 32) {
    v16h a;
#pragma unroll
    for (int v = 0; v < 8; ++v) {
      const int o0e = kc + kb16 + (v << 1);
      const int o1e = o0e + 1;
      const half_t* s0 =
          oT + ((size_t)(bp * 8 + (o0e >> 6)) * DH + (o0e & 63)) * NSPATIAL;
      const half_t* s1 =
          oT + ((size_t)(bp * 8 + (o1e >> 6)) * DH + (o1e & 63)) * NSPATIAL;
      a[2 * v]     = s0[nrow];
      a[2 * v + 1] = s1[nrow];
    }
#pragma unroll
    for (int j = 0; j < 4; ++j) {
      const int cq = c0 + j * 16 + col;
      const float* rp = Wout + (size_t)cq * INNER + kc + kb16;
      const float4 w0 = ((const float4*)rp)[0];
      const float4 w1 = ((const float4*)rp)[1];
      const float4 w2 = ((const float4*)rp)[2];
      const float4 w3 = ((const float4*)rp)[3];
      v16h b;
      b[0] = (half_t)w0.x; b[1] = (half_t)w0.y; b[2] = (half_t)w0.z; b[3] = (half_t)w0.w;
      b[4] = (half_t)w1.x; b[5] = (half_t)w1.y; b[6] = (half_t)w1.z; b[7] = (half_t)w1.w;
      b[8] = (half_t)w2.x; b[9] = (half_t)w2.y; b[10] = (half_t)w2.z; b[11] = (half_t)w2.w;
      b[12] = (half_t)w3.x; b[13] = (half_t)w3.y; b[14] = (half_t)w3.z; b[15] = (half_t)w3.w;
      acc[j] = wmma_f16(a, b, acc[j]);
    }
  }
#pragma unroll
  for (int j = 0; j < 4; ++j) {
    const int cq = c0 + j * 16 + col;
    const float bias = bout[cq];
    float* p = out + ((size_t)bp * QD + cq) * NSPATIAL + n0 + wave * 16 + rbase;
    float4 lo, hi;
    lo.x = acc[j][0] + bias; lo.y = acc[j][1] + bias;
    lo.z = acc[j][2] + bias; lo.w = acc[j][3] + bias;
    hi.x = acc[j][4] + bias; hi.y = acc[j][5] + bias;
    hi.z = acc[j][6] + bias; hi.w = acc[j][7] + bias;
    ((float4*)p)[0] = lo;
    ((float4*)p)[1] = hi;
  }
}

// ---------------------------------------------------------------------------
// Launch
// ---------------------------------------------------------------------------
extern "C" void kernel_launch(void* const* d_in, const int* in_sizes, int n_in,
                              void* d_out, int out_size, void* d_ws, size_t ws_size,
                              hipStream_t stream) {
  (void)in_sizes; (void)n_in; (void)out_size; (void)ws_size;
  const float* x    = (const float*)d_in[0];
  const float* Wq   = (const float*)d_in[1];
  const float* Wkv  = (const float*)d_in[2];
  const float* Wout = (const float*)d_in[3];
  const float* bout = (const float*)d_in[4];
  float* out = (float*)d_out;

  const size_t QKV_BYTES  = (size_t)NBP * GOUT * NSPATIAL * sizeof(half_t); // 96 MB
  const size_t SIM_BYTES  = (size_t)16 * 64 * 64 * sizeof(float);           // 256 KB
  const size_t ATTN_BYTES = (size_t)16 * 64 * 64 * sizeof(half_t);          // 128 KB
  char* ws = (char*)d_ws;
  half_t* qkv  = (half_t*)ws;
  float*  sim  = (float*)(ws + QKV_BYTES);
  half_t* attn = (half_t*)(ws + QKV_BYTES + SIM_BYTES);
  half_t* oT   = (half_t*)(ws + QKV_BYTES + SIM_BYTES + ATTN_BYTES);        // 32 MB

  qkv_proj_kernel<<<dim3(NSPATIAL / 128, GOUT / 64, NBP), 256, 0, stream>>>(
      x, Wq, Wkv, qkv);
  zero_sim_kernel<<<64, 256, 0, stream>>>(sim);
  sim_kernel<<<dim3(16, 16), 256, 0, stream>>>(qkv, sim);
  softmax_kernel<<<16 * 64, 64, 0, stream>>>(sim, attn);
  attnv_kernel<<<dim3(NSPATIAL / 128, 16), 256, 0, stream>>>(qkv, attn, oT);
  outproj_kernel<<<dim3(NSPATIAL / 128, QD / 64, NBP), 256, 0, stream>>>(
      oT, Wout, bout, out);
}
